// FraudLSTM_69904887709708
// MI455X (gfx1250) — matrix operand order repaired and checked by reading, not compiled, not run on hardware
//
#include <hip/hip_runtime.h>
#include <hip/hip_bf16.h>
#include <stdint.h>

// ---------------- problem sizes ----------------
#define B_SZ 1024
#define T_SZ 512
#define F_SZ 64
#define P_SZ 64
#define H_SZ 128
#define G_SZ 512   // 4*H

typedef __attribute__((ext_vector_type(16))) __bf16   v16bf;
typedef __attribute__((ext_vector_type(8)))  float    v8f;
typedef __attribute__((ext_vector_type(4)))  uint32_t v4u;

// ---------------- helpers ----------------
__device__ __forceinline__ uint16_t f2bf(float f) {
    union { __bf16 b; uint16_t u; } cv;
    cv.b = (__bf16)f;                 // native f32->bf16 convert on gfx1250
    return cv.u;
}
__device__ __forceinline__ float bf2f(uint16_t h) {
    return __uint_as_float(((uint32_t)h) << 16);
}
// Native v_tanh_f32 when available (CDNA5 TRANS op); one transcendental per activation.
__device__ __forceinline__ float tanh_hw(float x) {
#if __has_builtin(__builtin_amdgcn_tanhf)
    return __builtin_amdgcn_tanhf(x);
#else
    return 1.0f - 2.0f * __builtin_amdgcn_rcpf(__expf(2.0f * x) + 1.0f);
#endif
}
__device__ __forceinline__ float sigmoid_fast(float x) {
    return fmaf(0.5f, tanh_hw(0.5f * x), 0.5f);   // sigma(x) = (1+tanh(x/2))/2
}

// ---------------- kernel 1: fold projection into layer-0 input weights ----------------
// W_c[j][f] = sum_p W_ih0[j][p] * W_proj[p][f]      (512 x 64)
// bias0[j]  = W_ih0[j]·b_proj + b_ih0[j] + b_hh0[j]
// bias1[j]  = b_ih1[j] + b_hh1[j]
__global__ void prep_wc(const float* __restrict__ W_proj, const float* __restrict__ b_proj,
                        const float* __restrict__ W_ih0, const float* __restrict__ b_ih0,
                        const float* __restrict__ b_hh0, const float* __restrict__ b_ih1,
                        const float* __restrict__ b_hh1,
                        float* __restrict__ W_c, float* __restrict__ bias0,
                        float* __restrict__ bias1) {
    int j = blockIdx.x * blockDim.x + threadIdx.x;
    if (j >= G_SZ) return;
    for (int f = 0; f < F_SZ; ++f) {
        float acc = 0.f;
        for (int p = 0; p < P_SZ; ++p) acc += W_ih0[j * P_SZ + p] * W_proj[p * F_SZ + f];
        W_c[j * F_SZ + f] = acc;
    }
    float bacc = 0.f;
    for (int p = 0; p < P_SZ; ++p) bacc += W_ih0[j * P_SZ + p] * b_proj[p];
    bias0[j] = bacc + b_ih0[j] + b_hh0[j];
    bias1[j] = b_ih1[j] + b_hh1[j];
}

// ---------------- kernel 2: pre-swizzle weights into WMMA B-fragment layout ----------------
// B fragment, this code's contiguous k order (same as A): VGPR v, lanes 0-15 : K = 2v,2v+1 ; col = lane
//                                         lanes 16-31: K = 16+2v,17+2v ; col = lane-16
// linear dword index: ((kt*32 + ntile)*32 + lane)*8 + v
// Layer0: K = [ W_c^T (64 rows) ; W_hh0^T (128 rows) ]   -> KT=6  -> 49152 dwords
// Layer1: K = [ W_ih1^T (128)   ; W_hh1^T (128) ]        -> KT=8  -> 65536 dwords
__global__ void prep_bfrags(const float* __restrict__ W_c, const float* __restrict__ W_hh0,
                            const float* __restrict__ W_ih1, const float* __restrict__ W_hh1,
                            uint32_t* __restrict__ Bf0, uint32_t* __restrict__ Bf1) {
    const int D0 = 6 * 32 * 32 * 8;
    const int D1 = 8 * 32 * 32 * 8;
    int id = blockIdx.x * 256 + threadIdx.x;
    if (id >= D0 + D1) return;
    int buf = (id < D0) ? 0 : 1;
    int loc = buf ? (id - D0) : id;
    int v    = loc & 7;
    int lane = (loc >> 3) & 31;
    int nt   = (loc >> 8) & 31;
    int kt   = loc >> 13;
    int col  = nt * 16 + (lane & 15);            // gate index 0..511
    int rel0 = 2 * v + 16 * (lane >> 4);
    int K0   = kt * 32 + rel0;

    float e0, e1;
    if (buf == 0) {
        e0 = (K0     < 64) ? W_c[col * 64 + K0]       : W_hh0[col * 128 + (K0 - 64)];
        e1 = (K0 + 1 < 64) ? W_c[col * 64 + K0 + 1]   : W_hh0[col * 128 + (K0 + 1 - 64)];
    } else {
        e0 = (K0     < 128) ? W_ih1[col * 128 + K0]     : W_hh1[col * 128 + (K0 - 128)];
        e1 = (K0 + 1 < 128) ? W_ih1[col * 128 + K0 + 1] : W_hh1[col * 128 + (K0 + 1 - 128)];
    }
    uint32_t d = (uint32_t)f2bf(e0) | ((uint32_t)f2bf(e1) << 16);
    (buf ? Bf1 : Bf0)[loc] = d;
}

// ---------------- kernel 3: fused LSTM layer ----------------
// One block = 16 batch rows (1 WMMA row-tile), all T steps. 256 threads = 8 waves.
// Wave w owns hidden columns [16w,16w+16) of ALL four gates (ntiles w, w+8, w+16, w+24):
// the c,h elementwise update is fully in-register; cell state persists in VGPRs.
// Weight B-fragments are loop-invariant -> compiler keeps them register-resident.
// x and h tiles ping-pong in LDS: ONE barrier per timestep; next-step x staging
// overlaps WMMA compute.
template <int KTIN, bool IS_L0>
__global__ void __launch_bounds__(256)
lstm_kernel(const void* __restrict__ xin,       // L0: f32 (B,T,64); L1: bf16 (B,T,128)
            const uint32_t* __restrict__ Bf,    // pre-swizzled B fragments
            const float* __restrict__ bias,     // (512,)
            uint16_t* __restrict__ h_seq_out,   // L0: bf16 (B,T,128); L1: unused
            uint16_t* __restrict__ last_out)    // L1: bf16 (B,128);   L0: unused
{
    constexpr int KIN  = KTIN * 32;     // 64 or 128
    constexpr int KTOT = KTIN + 4;      // + H/32 recurrent k-tiles
    constexpr int ROWS = 16;
    constexpr int SX   = KIN + 2;       // padded LDS row strides (bank-conflict-free gathers)
    constexpr int SH   = H_SZ + 2;

    __shared__ __align__(16) uint16_t x_lds[2][ROWS * SX];   // ping-pong, row-major bf16
    __shared__ __align__(16) uint16_t h_lds[2][ROWS * SH];

    const int wave  = threadIdx.x >> 5;
    const int lane  = threadIdx.x & 31;
    const int l16   = lane & 15;
    const int khalf = lane >> 4;
    const int brow0 = blockIdx.x * ROWS;
    const int col   = wave * 16 + l16;          // this lane's hidden column

    // bias per gate tile (column-broadcast)
    float biasv[4];
#pragma unroll
    for (int g = 0; g < 4; ++g) biasv[g] = bias[(wave + 8 * g) * 16 + l16];

    // h0 = 0 (both buffers)
    for (int i = threadIdx.x; i < 2 * ROWS * SH; i += 256) h_lds[0][i] = 0;

    // cell state c0 = 0 (element (r + 8*khalf, col))
    float cst[8];
#pragma unroll
    for (int r = 0; r < 8; ++r) cst[r] = 0.f;

    // ---- stage a timestep's x tile into LDS (bf16) ----
    auto stage_x = [&](int t, uint16_t* xdst) {
        if (IS_L0) {
            const float* xf = (const float*)xin;
            for (int i = threadIdx.x; i < ROWS * 64; i += 256) {
                int r = i >> 6, f = i & 63;
                xdst[r * SX + f] = f2bf(xf[((size_t)(brow0 + r) * T_SZ + t) * 64 + f]);
            }
        } else {
            const uint32_t* xh = (const uint32_t*)xin;   // bf16 pairs, coalesced
            for (int i = threadIdx.x; i < ROWS * 64; i += 256) {
                int r = i >> 6, cp = i & 63;
                ((uint32_t*)&xdst[r * SX])[cp] =
                    xh[(size_t)(brow0 + r) * (T_SZ * 64) + (size_t)t * 64 + cp];
            }
        }
    };
    stage_x(0, &x_lds[0][0]);
    __syncthreads();

    // ---- one LSTM timestep; reads cur buffers, writes nxt buffers; one barrier ----
    auto step = [&](int t, const uint16_t* xcur, uint16_t* xnxt,
                    const uint16_t* hcur, uint16_t* hnxt) {
        // stage x_{t+1} into the other buffer (overlaps with WMMA below)
        if (t + 1 < T_SZ) stage_x(t + 1, xnxt);

        // accumulators: 4 gate-tiles, init with bias
        v8f C[4];
#pragma unroll
        for (int g = 0; g < 4; ++g) {
            v8f c;
#pragma unroll
            for (int r = 0; r < 8; ++r) c[r] = biasv[g];
            C[g] = c;
        }

        // K loop over [x ; h]
#pragma unroll
        for (int kt = 0; kt < KTOT; ++kt) {
            union { uint32_t u[8]; v16bf v; } af;
#pragma unroll
            for (int vv = 0; vv < 8; ++vv) {
                int rel = 2 * vv + 16 * khalf;
                uint32_t d;
                if (kt < KTIN) {
                    int K = kt * 32 + rel;
                    d = *(const uint32_t*)&xcur[l16 * SX + K];
                } else {
                    int K = (kt - KTIN) * 32 + rel;
                    d = *(const uint32_t*)&hcur[l16 * SH + K];
                }
                af.u[vv] = d;
            }
#pragma unroll
            for (int g = 0; g < 4; ++g) {
                int nt = wave + 8 * g;
                union { v4u q[2]; v16bf v; } bfr;
                const v4u* p = (const v4u*)(Bf + ((size_t)(kt * 32 + nt) * 32 + lane) * 8);
                bfr.q[0] = p[0];
                bfr.q[1] = p[1];
                C[g] = __builtin_amdgcn_wmma_f32_16x16x32_bf16(
                    false, af.v, false, bfr.v, (short)0, C[g], false, false);
            }
        }

        // gate activations + cell/hidden update; h -> LDS (nxt) and global (registers)
#pragma unroll
        for (int r = 0; r < 8; ++r) {
            float iv = sigmoid_fast(C[0][r]);
            float fv = sigmoid_fast(C[1][r]);
            float gv = tanh_hw(C[2][r]);
            float ov = sigmoid_fast(C[3][r]);
            float c  = fv * cst[r] + iv * gv;
            cst[r]   = c;
            uint16_t hb = f2bf(ov * tanh_hw(c));
            int row = r + 8 * khalf;
            hnxt[row * SH + col] = hb;
            if (IS_L0)
                h_seq_out[((size_t)(brow0 + row) * T_SZ + t) * H_SZ + col] = hb;
            else if (t == T_SZ - 1)
                last_out[(size_t)(brow0 + row) * H_SZ + col] = hb;
        }
        __syncthreads();   // single barrier: orders all ping-pong buffer hazards
    };

#pragma unroll 1
    for (int t = 0; t < T_SZ; t += 2) {
        step(t,     &x_lds[0][0], &x_lds[1][0], &h_lds[0][0], &h_lds[1][0]);
        step(t + 1, &x_lds[1][0], &x_lds[0][0], &h_lds[1][0], &h_lds[0][0]);
    }
}

// ---------------- kernel 4: MLP head ----------------
__global__ void head_kernel(const uint16_t* __restrict__ last, const float* __restrict__ W_h1,
                            const float* __restrict__ b_h1, const float* __restrict__ W_h2,
                            const float* __restrict__ b_h2, float* __restrict__ out) {
    __shared__ float sh[64];
    int b = blockIdx.x, j = threadIdx.x;
    float acc = b_h1[j];
    for (int k = 0; k < 128; ++k)
        acc += W_h1[j * 128 + k] * bf2f(last[(size_t)b * 128 + k]);
    acc = fmaxf(acc, 0.f);
    sh[j] = acc * W_h2[j];
    __syncthreads();
    if (j == 0) {
        float s = b_h2[0];
        for (int k = 0; k < 64; ++k) s += sh[k];
        out[b] = s;
    }
}

// ---------------- launcher ----------------
extern "C" void kernel_launch(void* const* d_in, const int* in_sizes, int n_in,
                              void* d_out, int out_size, void* d_ws, size_t ws_size,
                              hipStream_t stream) {
    (void)in_sizes; (void)n_in; (void)out_size; (void)ws_size;

    const float* x      = (const float*)d_in[0];
    /* d_in[1] = lengths (int64) — unused by reference output */
    const float* W_proj = (const float*)d_in[2];
    const float* b_proj = (const float*)d_in[3];
    const float* W_ih0  = (const float*)d_in[4];
    const float* W_hh0  = (const float*)d_in[5];
    const float* b_ih0  = (const float*)d_in[6];
    const float* b_hh0  = (const float*)d_in[7];
    const float* W_ih1  = (const float*)d_in[8];
    const float* W_hh1  = (const float*)d_in[9];
    const float* b_ih1  = (const float*)d_in[10];
    const float* b_hh1  = (const float*)d_in[11];
    const float* W_h1   = (const float*)d_in[12];
    const float* b_h1   = (const float*)d_in[13];
    const float* W_h2   = (const float*)d_in[14];
    const float* b_h2   = (const float*)d_in[15];

    char* ws = (char*)d_ws;
    float*    W_c    = (float*)(ws + 0);          // 512*64*4      = 131072
    float*    bias0  = (float*)(ws + 131072);     // 2048
    float*    bias1  = (float*)(ws + 133120);     // 2048
    uint32_t* Bf0    = (uint32_t*)(ws + 135168);  // 6*32*32*8*4   = 196608
    uint32_t* Bf1    = (uint32_t*)(ws + 331776);  // 8*32*32*8*4   = 262144
    uint16_t* last_h = (uint16_t*)(ws + 593920);  // 1024*128*2    = 262144
    uint16_t* h1seq  = (uint16_t*)(ws + 856064);  // 1024*512*128*2 = 134217728

    prep_wc<<<2, 256, 0, stream>>>(W_proj, b_proj, W_ih0, b_ih0, b_hh0, b_ih1, b_hh1,
                                   W_c, bias0, bias1);
    prep_bfrags<<<448, 256, 0, stream>>>(W_c, W_hh0, W_ih1, W_hh1, Bf0, Bf1);

    lstm_kernel<2, true ><<<B_SZ / 16, 256, 0, stream>>>(x,     Bf0, bias0, h1seq, nullptr);
    lstm_kernel<4, false><<<B_SZ / 16, 256, 0, stream>>>(h1seq, Bf1, bias1, nullptr, last_h);

    head_kernel<<<B_SZ, 64, 0, stream>>>(last_h, W_h1, b_h1, W_h2, b_h2, (float*)d_out);
}
